// PEB_73581379715506
// MI455X (gfx1250) — compile-verified, matrix operand order checked by reading, not run on hardware
//
#include <hip/hip_runtime.h>
#include <hip/hip_bf16.h>

// ---------------------------------------------------------------------------
// PEB forward for MI455X (gfx1250). All matmuls via v_wmma_f32_16x16x32_bf16,
// 16x64 (or wider) tiles per wave, contiguous bf16 fragment loads.
// ---------------------------------------------------------------------------

#define L_LEN 248
#define N_TOK 128
#define B_SZ  128
#define D_DIM 768
#define H_NUM 8
#define HD    96
#define HID_D 256
#define M_ROWS (B_SZ * N_TOK)          // 16384

typedef __attribute__((ext_vector_type(16))) __bf16 v16bf;
typedef __attribute__((ext_vector_type(8)))  float  v8f;

__device__ __forceinline__ v8f zero8() {
  v8f z = {0.f, 0.f, 0.f, 0.f, 0.f, 0.f, 0.f, 0.f};
  return z;
}

// K offset inside a 16x16x32 bf16 fragment for element i of this lane.
// lanes 0-15: K = {0..7, 16..23}; lanes 16-31: K = {8..15, 24..31}
__device__ __forceinline__ int wmma_koff(int lane, int i) {
  return ((i & 7) | ((i & 8) << 1)) + ((lane & 16) >> 1);
}

__device__ __forceinline__ v8f wmma_bf16(v16bf a, v16bf b, v8f c) {
  return __builtin_amdgcn_wmma_f32_16x16x32_bf16(false, a, false, b, (short)0, c,
                                                 false, false);
}

// A fragment from row-major fp32 [.. x lda]; per lane: 2 contiguous runs of 8
__device__ __forceinline__ v16bf fragA_f32(const float* A, size_t lda, int row0,
                                           int k0, int lane) {
  v16bf f;
  const float* p = A + (size_t)(row0 + (lane & 15)) * lda + k0;
#pragma unroll
  for (int i = 0; i < 16; ++i) f[i] = (__bf16)p[wmma_koff(lane, i)];
  return f;
}

// A/Bt fragment from row-major bf16 [.. x lda] (same lane layout both sides)
__device__ __forceinline__ v16bf fragA_bf(const __bf16* A, size_t lda, int row0,
                                          int k0, int lane) {
  v16bf f;
  const __bf16* p = A + (size_t)(row0 + (lane & 15)) * lda + k0;
#pragma unroll
  for (int i = 0; i < 16; ++i) f[i] = p[wmma_koff(lane, i)];
  return f;
}

// Bt fragment from bf16 [N x K], column index clamped (stores are guarded)
__device__ __forceinline__ v16bf fragBT_bf(const __bf16* Bt, size_t ldb, int k0,
                                           int col, int nmax, int lane) {
  int n = min(col + (lane & 15), nmax - 1);
  const __bf16* p = Bt + (size_t)n * ldb + k0;
  v16bf f;
#pragma unroll
  for (int i = 0; i < 16; ++i) f[i] = p[wmma_koff(lane, i)];
  return f;
}

// Bt fragment from fp32 [N x K] (contiguous in K), unguarded
__device__ __forceinline__ v16bf fragBT_f32(const float* Bt, size_t ldb, int k0,
                                            int col, int lane) {
  const float* p = Bt + (size_t)(col + (lane & 15)) * ldb + k0;
  v16bf f;
#pragma unroll
  for (int i = 0; i < 16; ++i) f[i] = (__bf16)p[wmma_koff(lane, i)];
  return f;
}

// ------------------------- block reductions --------------------------------
__device__ __forceinline__ float blk_sum(float v, float* sb) {
  int t = threadIdx.x, n = blockDim.x;
  sb[t] = v; __syncthreads();
  for (int s = n >> 1; s > 0; s >>= 1) {
    if (t < s) sb[t] += sb[t + s];
    __syncthreads();
  }
  float r = sb[0]; __syncthreads();
  return r;
}
__device__ __forceinline__ float blk_max(float v, float* sb) {
  int t = threadIdx.x, n = blockDim.x;
  sb[t] = v; __syncthreads();
  for (int s = n >> 1; s > 0; s >>= 1) {
    if (t < s) sb[t] = fmaxf(sb[t], sb[t + s]);
    __syncthreads();
  }
  float r = sb[0]; __syncthreads();
  return r;
}

// ------------------------- small prep kernels ------------------------------
__global__ void zero_kernel(float* p, size_t n) {
  size_t i = (size_t)blockIdx.x * blockDim.x + threadIdx.x;
  size_t st = (size_t)gridDim.x * blockDim.x;
  for (; i < n; i += st) p[i] = 0.f;
}

// scal[0]=kl_sum, [1]=rel_sum, [2]=n_valid, [3]=sum_b(cnt_b^2)
__global__ void stats_kernel(const int* word_pos, float* scal) {
  __shared__ float sc[128], sc2[128];
  int t = threadIdx.x;  // one thread per batch (128)
  int cnt = 0;
  for (int n = 0; n < N_TOK; ++n)
    if (word_pos[t * N_TOK + n] != -1) cnt++;
  sc[t] = (float)cnt;
  sc2[t] = (float)cnt * (float)cnt;
  __syncthreads();
  for (int s = 64; s > 0; s >>= 1) {
    if (t < s) { sc[t] += sc[t + s]; sc2[t] += sc2[t + s]; }
    __syncthreads();
  }
  if (t == 0) { scal[0] = 0.f; scal[1] = 0.f; scal[2] = sc[0]; scal[3] = sc2[0]; }
}

__global__ void qmask_kernel(const float* wf, const int* word_pos, float* qm,
                             size_t n) {
  size_t i = (size_t)blockIdx.x * blockDim.x + threadIdx.x;
  size_t st = (size_t)gridDim.x * blockDim.x;
  for (; i < n; i += st) {
    size_t r = i / D_DIM;
    qm[i] = (word_pos[r] != -1) ? wf[i] : 0.f;
  }
}

// dst[n*K + k] = (bf16) src[k*ld + c0 + n]   (weight transpose to [N][K] bf16)
__global__ void trw_kernel(const float* __restrict__ src, int ld, int c0, int K,
                           int N, __bf16* __restrict__ dst) {
  size_t total = (size_t)N * K;
  size_t i = (size_t)blockIdx.x * blockDim.x + threadIdx.x;
  size_t st = (size_t)gridDim.x * blockDim.x;
  for (; i < total; i += st) {
    int n = (int)(i / K), k = (int)(i % K);
    dst[i] = (__bf16)src[(size_t)k * ld + c0 + n];
  }
}

// vpT[d * M_ROWS + m] = (bf16) vp[m * D_DIM + d]
__global__ void vpt_kernel(const float* __restrict__ vp,
                           __bf16* __restrict__ vpT) {
  size_t total = (size_t)M_ROWS * D_DIM;
  size_t i = (size_t)blockIdx.x * blockDim.x + threadIdx.x;
  size_t st = (size_t)gridDim.x * blockDim.x;
  for (; i < total; i += st) {
    size_t m = i / D_DIM, d = i % D_DIM;
    vpT[d * (size_t)M_ROWS + m] = (__bf16)vp[i];
  }
}

// ------------------------- WMMA GEMM kernels -------------------------------
// ctx_anchor = gather(text_feat window) @ ctx_W + ctx_b, masked by vf.
// Wt = ctx_W transposed bf16 [768 x 2304]. 16x64 tile per wave.
__global__ void ctx_gemm_kernel(const float* __restrict__ tf,
                                const int* __restrict__ word_pos,
                                const __bf16* __restrict__ Wt,
                                const float* __restrict__ ctx_b,
                                float* __restrict__ out) {
  int lane = threadIdx.x & 31;
  int col0 = blockIdx.x * 64, row0 = blockIdx.y * 16;
  int r = row0 + (lane & 15);
  int b = r >> 7;
  int wpos = word_pos[r];
  int wp = min(max(wpos, 0), L_LEN - 1);
  v8f acc[4];
#pragma unroll
  for (int j = 0; j < 4; ++j) acc[j] = zero8();
  for (int w = 0; w < 3; ++w) {
    int p = min(max(wp - 1 + w, 0), L_LEN - 1);
    const float* Ap = tf + ((size_t)b * L_LEN + p) * D_DIM;
    for (int kk = 0; kk < D_DIM; kk += 32) {
      v16bf a;
#pragma unroll
      for (int i = 0; i < 16; ++i) a[i] = (__bf16)Ap[kk + wmma_koff(lane, i)];
      int k0 = w * D_DIM + kk;
#pragma unroll
      for (int j = 0; j < 4; ++j) {
        v16bf bf = fragBT_bf(Wt, 3 * D_DIM, k0, col0 + j * 16, D_DIM, lane);
        acc[j] = wmma_bf16(a, bf, acc[j]);
      }
    }
  }
#pragma unroll
  for (int j = 0; j < 4; ++j) {
    int n = col0 + j * 16 + (lane & 15);
    float bias = ctx_b[n];
#pragma unroll
    for (int rr = 0; rr < 8; ++rr) {
      int m = row0 + rr + ((lane & 16) >> 1);
      float vf = (word_pos[m] != -1) ? 1.f : 0.f;
      out[(size_t)m * D_DIM + n] = (acc[j][rr] + bias) * vf;
    }
  }
}

// Out[m,n] = act( A[m,:K] . Wt[n,:K] + bias[n] ); Wt bf16 [Nout x K]; 16x64 tile
__global__ void gemm_tbf_kernel(const float* __restrict__ A,
                                const __bf16* __restrict__ Wt,
                                const float* __restrict__ bias,
                                float* __restrict__ Out, int K, int Nout,
                                int ldo, int act) {
  int lane = threadIdx.x & 31;
  int col0 = blockIdx.x * 64, row0 = blockIdx.y * 16;
  v8f acc[4];
#pragma unroll
  for (int j = 0; j < 4; ++j) acc[j] = zero8();
  for (int k0 = 0; k0 < K; k0 += 32) {
    v16bf a = fragA_f32(A, (size_t)K, row0, k0, lane);
#pragma unroll
    for (int j = 0; j < 4; ++j) {
      v16bf b = fragBT_bf(Wt, (size_t)K, k0, col0 + j * 16, Nout, lane);
      acc[j] = wmma_bf16(a, b, acc[j]);
    }
  }
#pragma unroll
  for (int j = 0; j < 4; ++j) {
    int n = col0 + j * 16 + (lane & 15);
    if (n < Nout) {
      float bi = bias ? bias[n] : 0.f;
#pragma unroll
      for (int r = 0; r < 8; ++r) {
        int m = row0 + r + ((lane & 16) >> 1);
        float v = acc[j][r] + bi;
        if (act == 1) v = 0.5f * v * (1.f + erff(v * 0.70710678118654752f));
        Out[(size_t)m * ldo + n] = v;
      }
    }
  }
}

// logits[m,n] = (ropenbf[m,:] . cliprefbf[n,:]) / max(temp,1e-3); all bf16 in
__global__ void logits_kernel(const __bf16* __restrict__ Abf,
                              const __bf16* __restrict__ Bt,
                              float* __restrict__ Out,
                              const float* __restrict__ ptemp) {
  int lane = threadIdx.x & 31;
  int col0 = blockIdx.x * 64, row0 = blockIdx.y * 16;
  v8f acc[4];
#pragma unroll
  for (int j = 0; j < 4; ++j) acc[j] = zero8();
  for (int k0 = 0; k0 < D_DIM; k0 += 32) {
    v16bf a = fragA_bf(Abf, D_DIM, row0, k0, lane);
#pragma unroll
    for (int j = 0; j < 4; ++j) {
      v16bf b = fragBT_bf(Bt, D_DIM, k0, col0 + j * 16, L_LEN, lane);
      acc[j] = wmma_bf16(a, b, acc[j]);
    }
  }
  float scale = 1.f / fmaxf(ptemp[0], 0.001f);
#pragma unroll
  for (int j = 0; j < 4; ++j) {
    int n = col0 + j * 16 + (lane & 15);
    if (n < L_LEN) {
#pragma unroll
      for (int r = 0; r < 8; ++r) {
        int m = row0 + r + ((lane & 16) >> 1);
        Out[(size_t)m * L_LEN + n] = acc[j][r] * scale;
      }
    }
  }
}

// Full attention for one (b, h): QK^T -> softmax -> @V.  Scores in 64 KB LDS.
__global__ void attn_kernel(const float* __restrict__ qp,
                            const float* __restrict__ kp,
                            const __bf16* __restrict__ vpT,
                            float* __restrict__ ao) {
  __shared__ float S[128][128];
  int bh = blockIdx.x;
  int b = bh >> 3, h = bh & 7;
  int lane = threadIdx.x & 31, wid = threadIdx.x >> 5;
  size_t base = (size_t)b * N_TOK * D_DIM + (size_t)h * HD;
  const float* Q = qp + base;
  const float* Km = kp + base;

  // stage 1: each wave computes 2 row tiles x full 128 cols (8 accs)
  for (int rt2 = 0; rt2 < 2; ++rt2) {
    int rt = wid * 2 + rt2;
    v8f acc[8];
#pragma unroll
    for (int ct = 0; ct < 8; ++ct) acc[ct] = zero8();
    for (int k0 = 0; k0 < HD; k0 += 32) {
      v16bf a = fragA_f32(Q, D_DIM, rt * 16, k0, lane);
#pragma unroll
      for (int ct = 0; ct < 8; ++ct) {
        v16bf bf = fragBT_f32(Km, D_DIM, k0, ct * 16, lane);
        acc[ct] = wmma_bf16(a, bf, acc[ct]);
      }
    }
    const float rs = 0.10206207261596577f;  // 1/sqrt(96)
#pragma unroll
    for (int ct = 0; ct < 8; ++ct) {
#pragma unroll
      for (int r = 0; r < 8; ++r) {
        int m = rt * 16 + r + ((lane & 16) >> 1);
        int n = ct * 16 + (lane & 15);
        S[m][n] = acc[ct][r] * rs;
      }
    }
  }
  __syncthreads();

  // softmax rows (one row per thread, 128 threads)
  {
    int q = threadIdx.x;
    float mx = -1e30f;
    for (int k = 0; k < 128; ++k) mx = fmaxf(mx, S[q][k]);
    float sum = 0.f;
    for (int k = 0; k < 128; ++k) {
      float e = __expf(S[q][k] - mx);
      S[q][k] = e;
      sum += e;
    }
    float inv = 1.f / sum;
    for (int k = 0; k < 128; ++k) S[q][k] *= inv;
  }
  __syncthreads();

  // stage 2: O = attn @ V; B from transposed bf16 vpT (contiguous in K)
  const __bf16* Vt = vpT + (size_t)h * HD * M_ROWS + (size_t)b * N_TOK;
  for (int rt2 = 0; rt2 < 2; ++rt2) {
    int rt = wid * 2 + rt2;
    v8f acc[6];
#pragma unroll
    for (int ct = 0; ct < 6; ++ct) acc[ct] = zero8();
    for (int k0 = 0; k0 < 128; k0 += 32) {
      v16bf a;
      int mrow = rt * 16 + (lane & 15);
#pragma unroll
      for (int i = 0; i < 16; ++i)
        a[i] = (__bf16)S[mrow][k0 + wmma_koff(lane, i)];
#pragma unroll
      for (int ct = 0; ct < 6; ++ct) {
        const __bf16* p = Vt + (size_t)(ct * 16 + (lane & 15)) * M_ROWS + k0;
        v16bf bf;
#pragma unroll
        for (int i = 0; i < 16; ++i) bf[i] = p[wmma_koff(lane, i)];
        acc[ct] = wmma_bf16(a, bf, acc[ct]);
      }
    }
#pragma unroll
    for (int ct = 0; ct < 6; ++ct) {
#pragma unroll
      for (int r = 0; r < 8; ++r) {
        int m = rt * 16 + r + ((lane & 16) >> 1);
        int n = ct * 16 + (lane & 15);
        ao[base + (size_t)m * D_DIM + n] = acc[ct][r];
      }
    }
  }
}

// ------------------------- normalization kernels ---------------------------
__global__ void ln_residual_kernel(const float* __restrict__ q,
                                   const float* __restrict__ ao,
                                   const int* __restrict__ word_pos,
                                   const float* __restrict__ g,
                                   const float* __restrict__ be,
                                   float* __restrict__ x) {
  __shared__ float sb[256];
  int r = blockIdx.x, t = threadIdx.x;
  float vf = (word_pos[r] != -1) ? 1.f : 0.f;
  float vals[3];
  float s = 0.f;
#pragma unroll
  for (int i = 0; i < 3; ++i) {
    int d = t + i * 256;
    float v = q[(size_t)r * D_DIM + d] + ao[(size_t)r * D_DIM + d];
    vals[i] = v;
    s += v;
  }
  float mean = blk_sum(s, sb) * (1.f / D_DIM);
  float vs = 0.f;
#pragma unroll
  for (int i = 0; i < 3; ++i) {
    float d0 = vals[i] - mean;
    vs += d0 * d0;
  }
  float var = blk_sum(vs, sb) * (1.f / D_DIM);
  float inv = rsqrtf(var + 1e-5f);
#pragma unroll
  for (int i = 0; i < 3; ++i) {
    int d = t + i * 256;
    x[(size_t)r * D_DIM + d] = ((vals[i] - mean) * inv * g[d] + be[d]) * vf;
  }
}

__global__ void ln_hid_kernel(const float* __restrict__ g,
                              const float* __restrict__ be,
                              float* __restrict__ h) {
  __shared__ float sb[256];
  int r = blockIdx.x, t = threadIdx.x;
  float v = h[(size_t)r * HID_D + t];
  float mean = blk_sum(v, sb) * (1.f / HID_D);
  float d = v - mean;
  float var = blk_sum(d * d, sb) * (1.f / HID_D);
  h[(size_t)r * HID_D + t] = d * rsqrtf(var + 1e-5f) * g[t] + be[t];
}

// Y_bf16 = X / max(||X||, 1e-12) per row of D=768
__global__ void rownorm_bf_kernel(const float* __restrict__ X,
                                  __bf16* __restrict__ Y) {
  __shared__ float sb[256];
  int r = blockIdx.x, t = threadIdx.x;
  float vals[3];
  float s = 0.f;
#pragma unroll
  for (int i = 0; i < 3; ++i) {
    int d = t + i * 256;
    float v = X[(size_t)r * D_DIM + d];
    vals[i] = v;
    s += v * v;
  }
  float nrm = sqrtf(blk_sum(s, sb));
  float inv = 1.f / fmaxf(nrm, 1e-12f);
#pragma unroll
  for (int i = 0; i < 3; ++i) {
    int d = t + i * 256;
    Y[(size_t)r * D_DIM + d] = (__bf16)(vals[i] * inv);
  }
}

// ------------------------- losses + scatter --------------------------------
__global__ void kl_kernel(const float* __restrict__ logits,
                          const int* __restrict__ word_pos,
                          float* __restrict__ kl_sum) {
  __shared__ float sb[256];
  int r = blockIdx.x;
  int wpos = word_pos[r];
  if (wpos == -1) return;  // uniform across block
  int t = threadIdx.x;
  int wp = min(max(wpos, 0), L_LEN - 1);
  float lg = (t < L_LEN) ? logits[(size_t)r * L_LEN + t] : -1e30f;
  float mx = blk_max(lg, sb);
  float e = (t < L_LEN) ? __expf(lg - mx) : 0.f;
  float se = blk_sum(e, sb);
  float lse = mx + logf(se);
  float tun = 0.f;
  if (t < L_LEN) {
    float d = (float)t - (float)wp;
    tun = __expf(-d * d * 0.125f);  // 1/(2*sigma^2), sigma=2
  }
  float tsum = blk_sum(tun, sb);
  float term = 0.f;
  if (t < L_LEN) {
    float targ = tun / (tsum + 1e-12f);
    if (targ > 0.f) term = targ * (logf(targ) - (lg - lse));
  }
  float s = blk_sum(term, sb);
  if (t == 0) atomicAdd(kl_sum, s);
}

// rel loss: 16x64 WMMA tile of rope_n @ rope_n^T within batch, then sq-diff
__global__ void rel_kernel(const __bf16* __restrict__ ropenbf,
                           const int* __restrict__ word_pos,
                           float* __restrict__ rel_sum) {
  int lane = threadIdx.x & 31;
  int b = blockIdx.z;
  int it = blockIdx.x * 16, j0 = blockIdx.y * 64;
  const __bf16* Ab = ropenbf + (size_t)b * N_TOK * D_DIM;
  v8f acc[4];
#pragma unroll
  for (int j = 0; j < 4; ++j) acc[j] = zero8();
  for (int k0 = 0; k0 < D_DIM; k0 += 32) {
    v16bf a = fragA_bf(Ab, D_DIM, it, k0, lane);
#pragma unroll
    for (int j = 0; j < 4; ++j) {
      v16bf bb = fragA_bf(Ab, D_DIM, j0 + j * 16, k0, lane);
      acc[j] = wmma_bf16(a, bb, acc[j]);
    }
  }
  float lsum = 0.f;
#pragma unroll
  for (int j = 0; j < 4; ++j) {
    int jj = j0 + j * 16 + (lane & 15);
    int wj = word_pos[b * N_TOK + jj];
    float vj = (wj != -1) ? 1.f : 0.f;
    float wjf = (float)min(max(wj, 0), L_LEN - 1);
#pragma unroll
    for (int r = 0; r < 8; ++r) {
      int i = it + r + ((lane & 16) >> 1);
      int wi = word_pos[b * N_TOK + i];
      float vi = (wi != -1) ? 1.f : 0.f;
      float wif = (float)min(max(wi, 0), L_LEN - 1);
      float s01 = (acc[j][r] + 1.f) * 0.5f;
      float tgt = 1.f - fabsf(wif - wjf) * (1.f / (float)L_LEN);
      float vm2 = vi * vj;
      float d = (s01 - tgt) * vm2;
      lsum += d * d;
    }
  }
  for (int off = 16; off > 0; off >>= 1) lsum += __shfl_down(lsum, off, 32);
  if (lane == 0) atomicAdd(rel_sum, lsum);
}

__global__ void scatter_kernel(const float* __restrict__ rope,
                               const float* __restrict__ pos_emb,
                               const int* __restrict__ word_pos,
                               float* __restrict__ emb) {
  int r = blockIdx.x, t = threadIdx.x;
  int wpos = word_pos[r];
  if (wpos == -1) return;
  int wp = min(max(wpos, 0), L_LEN - 1);
  int b = r >> 7;
#pragma unroll
  for (int i = 0; i < 3; ++i) {
    int d = t + i * 256;
    emb[((size_t)b * L_LEN + wp) * D_DIM + d] =
        0.8f * pos_emb[(size_t)wp * D_DIM + d] +
        0.2f * rope[(size_t)r * D_DIM + d];
  }
}

__global__ void finalize_kernel(const float* __restrict__ scal,
                                float* __restrict__ out_scalars) {
  float kl = scal[0] / (scal[2] + 1e-12f);
  float rel = scal[1] / (scal[3] + 1e-12f);
  out_scalars[0] = kl + 0.5f * rel;
  out_scalars[1] = kl;
  out_scalars[2] = rel;
}

// ---------------------------------------------------------------------------
extern "C" void kernel_launch(void* const* d_in, const int* in_sizes, int n_in,
                              void* d_out, int out_size, void* d_ws,
                              size_t ws_size, hipStream_t stream) {
  (void)in_sizes; (void)n_in; (void)out_size; (void)ws_size;

  const float* word_feat = (const float*)d_in[0];
  const float* text_feat = (const float*)d_in[1];
  const int*   word_pos  = (const int*)d_in[2];
  const float* ctx_W = (const float*)d_in[3];
  const float* ctx_b = (const float*)d_in[4];
  const float* in_W  = (const float*)d_in[5];
  const float* in_b  = (const float*)d_in[6];
  const float* out_W = (const float*)d_in[7];
  const float* out_b = (const float*)d_in[8];
  const float* cn_g  = (const float*)d_in[9];
  const float* cn_b  = (const float*)d_in[10];
  const float* r1_W  = (const float*)d_in[11];
  const float* r1_b  = (const float*)d_in[12];
  const float* ln_g  = (const float*)d_in[13];
  const float* ln_b  = (const float*)d_in[14];
  const float* r2_W  = (const float*)d_in[15];
  const float* r2_b  = (const float*)d_in[16];
  const float* pos_emb  = (const float*)d_in[17];
  const float* pos_temp = (const float*)d_in[18];

  float* out = (float*)d_out;
  const size_t EMB_N = (size_t)B_SZ * L_LEN * D_DIM;  // 24,379,392

  // ---------------- workspace layout (~305 MB with reuse) ----------------
  const size_t MD = (size_t)M_ROWS * D_DIM;  // 12,582,912
  float* ws = (float*)d_ws;
  float* qm   = ws;          // masked word_feat (kept for residual)
  float* ctxa = qm + MD;     // ctx_anchor, later reused as ao
  float* qp   = ctxa + MD;   // q-proj, later reused as x
  float* kp   = qp + MD;     // k-proj, later reused as rope (fp32)
  float* vp   = kp + MD;     // v-proj, later attn_out; later bf16 rope_n alias
  float* h1   = vp + MD;     // MLP hidden (16384x256), later logits (x248)
  float* scal = h1 + (size_t)M_ROWS * HID_D;  // 4 accum scalars
  __bf16* wt_ctx = (__bf16*)(scal + 8);                 // 768 x 2304
  __bf16* wt_q   = wt_ctx + (size_t)D_DIM * 3 * D_DIM;  // 768 x 768
  __bf16* wt_k   = wt_q + (size_t)D_DIM * D_DIM;
  __bf16* wt_v   = wt_k + (size_t)D_DIM * D_DIM;
  __bf16* wt_o   = wt_v + (size_t)D_DIM * D_DIM;
  __bf16* wt_r1  = wt_o + (size_t)D_DIM * D_DIM;        // 256 x 768
  __bf16* wt_r2  = wt_r1 + (size_t)HID_D * D_DIM;       // 768 x 256
  __bf16* cliprefbf = wt_r2 + (size_t)D_DIM * HID_D;    // 248 x 768
  __bf16* vpT = cliprefbf + (size_t)L_LEN * D_DIM;      // 768 x 16384
  __bf16* ropenbf = (__bf16*)vp;                        // 16384 x 768 (alias)

  const dim3 g64(12, 1024);   // Nout = 768, 16x64 tiles
  const dim3 g64h(4, 1024);   // Nout = 256 / 248

  // 0) init output / stats / masked q; transpose weights to bf16 [N][K]
  zero_kernel<<<2048, 256, 0, stream>>>(out, EMB_N);
  stats_kernel<<<1, 128, 0, stream>>>(word_pos, scal);
  qmask_kernel<<<2048, 256, 0, stream>>>(word_feat, word_pos, qm, MD);
  trw_kernel<<<1024, 256, 0, stream>>>(ctx_W, D_DIM, 0, 3 * D_DIM, D_DIM, wt_ctx);
  trw_kernel<<<512, 256, 0, stream>>>(in_W, 3 * D_DIM, 0,         D_DIM, D_DIM, wt_q);
  trw_kernel<<<512, 256, 0, stream>>>(in_W, 3 * D_DIM, D_DIM,     D_DIM, D_DIM, wt_k);
  trw_kernel<<<512, 256, 0, stream>>>(in_W, 3 * D_DIM, 2 * D_DIM, D_DIM, D_DIM, wt_v);
  trw_kernel<<<512, 256, 0, stream>>>(out_W, D_DIM, 0, D_DIM, D_DIM, wt_o);
  trw_kernel<<<256, 256, 0, stream>>>(r1_W, HID_D, 0, D_DIM, HID_D, wt_r1);
  trw_kernel<<<256, 256, 0, stream>>>(r2_W, D_DIM, 0, HID_D, D_DIM, wt_r2);
  rownorm_bf_kernel<<<L_LEN, 256, 0, stream>>>(pos_emb, cliprefbf);

  // 1) ctx_anchor = gather-window GEMM (K = 2304)
  ctx_gemm_kernel<<<g64, 32, 0, stream>>>(text_feat, word_pos, wt_ctx, ctx_b, ctxa);

  // 2) Q/K/V projections
  gemm_tbf_kernel<<<g64, 32, 0, stream>>>(qm,   wt_q, in_b,        qp, D_DIM, D_DIM, D_DIM, 0);
  gemm_tbf_kernel<<<g64, 32, 0, stream>>>(ctxa, wt_k, in_b + 768,  kp, D_DIM, D_DIM, D_DIM, 0);
  gemm_tbf_kernel<<<g64, 32, 0, stream>>>(ctxa, wt_v, in_b + 1536, vp, D_DIM, D_DIM, D_DIM, 0);
  vpt_kernel<<<2048, 256, 0, stream>>>(vp, vpT);

  // 3) attention per (b,h); ao reuses ctx_anchor buffer
  attn_kernel<<<B_SZ * H_NUM, 128, 0, stream>>>(qp, kp, vpT, ctxa);

  // 4) out-projection (into vp), then residual + LN (x into qp)
  gemm_tbf_kernel<<<g64, 32, 0, stream>>>(ctxa, wt_o, out_b, vp, D_DIM, D_DIM, D_DIM, 0);
  ln_residual_kernel<<<M_ROWS, 256, 0, stream>>>(qm, vp, word_pos, cn_g, cn_b, qp);

  // 5) MLP: gelu(x @ r1_W + b), LN(hid), @ r2_W + b -> rope (kp buffer)
  gemm_tbf_kernel<<<g64h, 32, 0, stream>>>(qp, wt_r1, r1_b, h1, D_DIM, HID_D, HID_D, 1);
  ln_hid_kernel<<<M_ROWS, 256, 0, stream>>>(ln_g, ln_b, h1);
  gemm_tbf_kernel<<<g64, 32, 0, stream>>>(h1, wt_r2, r2_b, kp, HID_D, D_DIM, D_DIM, 0);

  // 6) rope_n as bf16 (vp region free now)
  rownorm_bf_kernel<<<M_ROWS, 256, 0, stream>>>(kp, ropenbf);

  // 7) logits = rope_n @ clip_ref_n^T / temp  (into h1 buffer)
  logits_kernel<<<g64h, 32, 0, stream>>>(ropenbf, cliprefbf, h1, pos_temp);

  // 8) output embedding scatter (uses un-normalized rope in kp)
  scatter_kernel<<<M_ROWS, 256, 0, stream>>>(kp, pos_emb, word_pos, out);

  // 9) losses
  kl_kernel<<<M_ROWS, 256, 0, stream>>>(h1, word_pos, scal + 0);
  rel_kernel<<<dim3(8, 2, B_SZ), 32, 0, stream>>>(ropenbf, word_pos, scal + 1);
  finalize_kernel<<<1, 1, 0, stream>>>(scal, out + EMB_N);
}
